// TernaryConv2d_68152541053472
// MI455X (gfx1250) — matrix operand order repaired and checked by reading, not compiled, not run on hardware
//
#include <hip/hip_runtime.h>
#include <hip/hip_bf16.h>

typedef __attribute__((ext_vector_type(8)))  __bf16 v8bf;
typedef __attribute__((ext_vector_type(16))) __bf16 v16bf;
typedef __attribute__((ext_vector_type(8)))  float  v8f;
typedef int v4i_async __attribute__((vector_size(16)));

// d_ws layout:
//   [0, 1024)            alpha f32[256]
//   [1024, ~1.2MB)       packed ternary bf16 [9][256][256]   (tap, k, c)
//   [2MB, ~57MB)         x as bf16 NHWC with zero halo: [32][58][58][256]
#define WQ_BYTE_OFF  1024
#define XBF_BYTE_OFF (1u << 21)

// ---- CDNA5 async global->LDS copy (ASYNCcnt) with safe fallback -----------
#if __has_builtin(__builtin_amdgcn_global_load_async_to_lds_b128)
#define ASYNC_CP16(dst, src)                                              \
  __builtin_amdgcn_global_load_async_to_lds_b128(                         \
      (__attribute__((address_space(1))) v4i_async*)(src),                \
      (__attribute__((address_space(3))) v4i_async*)(dst), 0, 0)
#define HAVE_ASYNC_LDS 1
#else
#define ASYNC_CP16(dst, src) (*(uint4*)(dst) = *(const uint4*)(src))
#define HAVE_ASYNC_LDS 0
#endif

__device__ __forceinline__ void async_wait_all() {
#if HAVE_ASYNC_LDS
#if __has_builtin(__builtin_amdgcn_s_wait_asynccnt)
  __builtin_amdgcn_s_wait_asynccnt(0);
#else
  asm volatile("s_wait_asynccnt 0x0" ::: "memory");
#endif
#endif
}

// ---------------------------------------------------------------------------
// Kernel 1: ternary quantize + pack weights.
// One block per output channel o. weight is OIHW: idx = o*2304 + c*9 + tap.
// Packed layout: wq[tap*65536 + o*256 + c]  (c contiguous for GEMM A tiles).
// ---------------------------------------------------------------------------
__global__ __launch_bounds__(256) void tq_pack_kernel(
    const float* __restrict__ w, float* __restrict__ alpha,
    __bf16* __restrict__ wq) {
  const int o = blockIdx.x;
  const int t = threadIdx.x;
  const float* wo = w + o * 2304;

  __shared__ float redf[256];
  __shared__ int   redi[256];

  float vals[9];
  float s = 0.f;
#pragma unroll
  for (int j = 0; j < 9; ++j) {
    vals[j] = wo[t + 256 * j];
    s += fabsf(vals[j]);
  }
  redf[t] = s;
  __syncthreads();
  for (int off = 128; off; off >>= 1) {
    if (t < off) redf[t] += redf[t + off];
    __syncthreads();
  }
  const float delta = 0.7f * redf[0] / 2304.f;
  __syncthreads();

  float tern[9];
  float sm = 0.f;
  int cnt = 0;
#pragma unroll
  for (int j = 0; j < 9; ++j) {
    const float v = vals[j];
    const float tv = (v > delta) ? 1.f : ((v < -delta) ? -1.f : 0.f);
    tern[j] = tv;
    if (tv != 0.f) { cnt++; sm += fabsf(v); }
  }
  redf[t] = sm;
  redi[t] = cnt;
  __syncthreads();
  for (int off = 128; off; off >>= 1) {
    if (t < off) { redf[t] += redf[t + off]; redi[t] += redi[t + off]; }
    __syncthreads();
  }
  if (t == 0) {
    const int c = redi[0] > 0 ? redi[0] : 1;
    alpha[o] = redf[0] / (float)c;
  }
#pragma unroll
  for (int j = 0; j < 9; ++j) {
    const int flat = t + 256 * j;   // == c*9 + tap
    const int c = flat / 9;
    const int tap = flat % 9;
    wq[tap * 65536 + o * 256 + c] = (__bf16)tern[j];
  }
}

// ---------------------------------------------------------------------------
// Kernel 2: NCHW f32 -> padded NHWC bf16 ([32][58][58][256], zero halo).
// ---------------------------------------------------------------------------
__global__ __launch_bounds__(256) void x_pad_nhwc_kernel(
    const float* __restrict__ x, __bf16* __restrict__ xbf) {
  const unsigned idx = blockIdx.x * 256u + threadIdx.x;  // < 32*58*58*256
  const int c  = idx & 255;
  const int wp = (idx >> 8) % 58;
  const int hp = ((idx >> 8) / 58) % 58;
  const int n  = (idx >> 8) / (58 * 58);
  float v = 0.f;
  if (hp >= 1 && hp <= 56 && wp >= 1 && wp <= 56)
    v = x[((n * 256 + c) * 56 + (hp - 1)) * 56 + (wp - 1)];
  xbf[idx] = (__bf16)v;
}

// ---------------------------------------------------------------------------
// Kernel 3: implicit-GEMM conv via v_wmma_f32_16x16x32_bf16.
// Block tile: M=64 out-channels x N=128 positions (4 rows x 32 cols).
// 8 waves: 2 M-groups x 4 N-groups, each wave = 2x2 WMMA 16x16 tiles.
// K loop: 8 chunks of 32 input channels x 9 taps.
// Double-buffered LDS, GLOBAL_LOAD_ASYNC_TO_LDS_B128 prefetch of chunk i+1
// overlapped with WMMA compute of chunk i; one barrier per iteration.
// ---------------------------------------------------------------------------
__global__ __launch_bounds__(256) void ternary_conv_kernel(
    const __bf16* __restrict__ xbf, const __bf16* __restrict__ wq,
    const float* __restrict__ alpha, float* __restrict__ out) {
  // Double-buffered LDS: weights [9][64 k][32 c], x tile [6 rows][34 cols][32 c]
  __shared__ __align__(16) __bf16 Wt[2][9 * 64 * 32];   // 2 x 36864 B
  __shared__ __align__(16) __bf16 Xt[2][6 * 34 * 32];   // 2 x 13056 B

  const int ct = blockIdx.x & 1;        // col tile: w0 = 0 or 24 (overlapped)
  const int rt = blockIdx.x >> 1;       // row tile: 0..13
  const int n  = blockIdx.y;            // image
  const int k0 = blockIdx.z * 64;       // out-channel tile
  const int h0 = rt * 4;
  const int w0 = ct * 24;               // tile1 covers w 24..55, stores 32..55

  const int t     = threadIdx.x;
  const int lane  = t & 31;
  const int wave  = t >> 5;
  const int waveM = wave & 1;           // 0..1 -> M tiles {2*waveM, 2*waveM+1}
  const int waveN = wave >> 1;          // 0..3 -> N tiles {2*waveN, 2*waveN+1}
  const int m     = lane & 15;          // row / column within 16x16 tile
  const int half  = lane >> 4;          // K-half select per WMMA operand layout

  // padded NHWC base for this image; halo offset +1 is absorbed: original
  // pixel (h0-1+row, w0-1+col) == padded (h0+row, w0+col)
  const __bf16* xn = xbf + (size_t)n * 58 * 58 * 256;

  // Issue the async staging batch for input-channel chunk c0 into buffer buf.
  auto stage = [&](int buf, int c0) {
    // x tile: 204 pixels x 32 ch = 816 x 16B chunks, no bounds checks
    for (int idx = t; idx < 6 * 34 * 4; idx += 256) {
      const int pix = idx >> 2;                         // 0..203
      const int cc  = (idx & 3) * 8;
      const int row = pix / 34;
      const int col = pix % 34;
      ASYNC_CP16(&Xt[buf][pix * 32 + cc],
                 &xn[(((h0 + row) * 58) + (w0 + col)) * 256 + c0 + cc]);
    }
    // all 9 weight taps for this (k0, c0) tile: 2304 x 16B chunks
    for (int ci = t; ci < 9 * 64 * 4; ci += 256) {
      const int tap = ci >> 8;
      const int rem = ci & 255;
      const int k  = rem >> 2;
      const int cc = (rem & 3) * 8;
      ASYNC_CP16(&Wt[buf][ci * 8],
                 &wq[tap * 65536 + (k0 + k) * 256 + c0 + cc]);
    }
  };

  v8f acc[2][2] = {};

  stage(0, 0);
  for (int it = 0; it < 8; ++it) {
    // Complete the in-flight batch for buffer it&1, make it visible to all
    // waves. The same barrier also guarantees everyone finished reading the
    // buffer we are about to refill (last touched in iteration it-1).
    async_wait_all();
    __syncthreads();
    if (it < 7) stage((it + 1) & 1, (it + 1) * 32);   // overlap with compute

    const __bf16* Wb = Wt[it & 1];
    const __bf16* Xb = Xt[it & 1];

#pragma unroll
    for (int tap = 0; tap < 9; ++tap) {
      const int r  = tap / 3;
      const int s0 = tap % 3;

      // A fragments (weights): lane m = M row; K chunks at half*8 and 16+half*8
      v16bf a[2];
#pragma unroll
      for (int mi = 0; mi < 2; ++mi) {
        const int krow = (waveM * 2 + mi) * 16 + m;
        const __bf16* base = &Wb[tap * 2048 + krow * 32];
        const v8bf a0 = *(const v8bf*)(base + half * 8);
        const v8bf a1 = *(const v8bf*)(base + 16 + half * 8);
        a[mi] = __builtin_shufflevector(a0, a1, 0, 1, 2, 3, 4, 5, 6, 7, 8, 9,
                                        10, 11, 12, 13, 14, 15);
      }
      // B fragments (x): lane m = N column; K chunks at half*8 and 16+half*8
      v16bf b[2];
#pragma unroll
      for (int ni = 0; ni < 2; ++ni) {
        const int pos  = (waveN * 2 + ni) * 16 + m;  // 0..127 in tile
        const int orow = pos >> 5;
        const int ocol = pos & 31;
        const __bf16* base =
            &Xb[((orow + r) * 34 + (ocol + s0)) * 32];
        const v8bf b0 = *(const v8bf*)(base + half * 8);
        const v8bf b1 = *(const v8bf*)(base + 16 + half * 8);
        b[ni] = __builtin_shufflevector(b0, b1, 0, 1, 2, 3, 4, 5, 6, 7, 8, 9,
                                        10, 11, 12, 13, 14, 15);
      }
#pragma unroll
      for (int mi = 0; mi < 2; ++mi)
#pragma unroll
        for (int ni = 0; ni < 2; ++ni)
          acc[mi][ni] = __builtin_amdgcn_wmma_f32_16x16x32_bf16(
              false, a[mi], false, b[ni], (short)0, acc[mi][ni], false, false);
    }
  }

  // ---- epilogue: D element (vgpr i, lane l): M = i + 8*(l>>4), N = l&15
  const int ocol_min = w0 ? 8 : 0;   // overlapped col tile stores only w>=32
#pragma unroll
  for (int mi = 0; mi < 2; ++mi) {
    const int ktile = (waveM * 2 + mi) * 16;
#pragma unroll
    for (int ni = 0; ni < 2; ++ni) {
      const int pos  = (waveN * 2 + ni) * 16 + m;
      const int orow = pos >> 5;
      const int ocol = pos & 31;
      const int h = h0 + orow;
      const int w = w0 + ocol;
      if (ocol >= ocol_min) {
#pragma unroll
        for (int i = 0; i < 8; ++i) {
          const int k = k0 + ktile + i + 8 * half;
          out[((n * 256 + k) * 56 + h) * 56 + w] = alpha[k] * acc[mi][ni][i];
        }
      }
    }
  }
}

// ---------------------------------------------------------------------------
extern "C" void kernel_launch(void* const* d_in, const int* in_sizes, int n_in,
                              void* d_out, int out_size, void* d_ws,
                              size_t ws_size, hipStream_t stream) {
  const float* x = (const float*)d_in[0];
  const float* w = (const float*)d_in[1];
  float*  alpha = (float*)d_ws;
  __bf16* wq    = (__bf16*)((char*)d_ws + WQ_BYTE_OFF);
  __bf16* xbf   = (__bf16*)((char*)d_ws + XBF_BYTE_OFF);
  float*  out   = (float*)d_out;

  // 1) quantize + pack weights: 256 blocks (one per output channel)
  tq_pack_kernel<<<256, 256, 0, stream>>>(w, alpha, wq);

  // 2) x: NCHW f32 -> padded NHWC bf16 (one streaming pass)
  const unsigned xtotal = 32u * 58 * 58 * 256;
  x_pad_nhwc_kernel<<<xtotal / 256, 256, 0, stream>>>(x, xbf);

  // 3) conv: grid = (2 col-tiles * 14 row-tiles, 32 images, 4 k-tiles)
  dim3 grid(28, 32, 4);
  ternary_conv_kernel<<<grid, 256, 0, stream>>>(xbf, wq, alpha, out);
}
